// ProtoConv2d_8452495638613
// MI455X (gfx1250) — matrix operand order repaired and checked by reading, not compiled, not run on hardware
//
#include <hip/hip_runtime.h>
#include <hip/hip_bf16.h>

// ---------------------------------------------------------------------------
// ProtoConv2d for MI455X (gfx1250, wave32, WMMA bf16 w/ f32 accumulate)
//
// Shapes: B=16, C_IN=64, H=W=64, C_OUT=128, K=3, P=1, NC=512, PS=576
//   N = B*H*W = 65536 patch rows, L = 4096 pixels per batch
// Pipeline:
//   K1 im2col      : x -> flat32 [N,576] f32, flatB [N,576] bf16, rn [N]
//   K2 prep_centers: cc -> ccB [512,576] bf16, ccT [576,512] bf16, cn [512]
//   K2w prep_w     : weight -> W2 [128,576] bf16
//   K3 gemm_fused  : per 16-row tile: WMMA GEMM1 (K=576) -> sqrt/softmax ->
//                    WMMA GEMM2 (K=512) -> blend -> final [N,576] f32
//   K4 conv_out    : final viewed [B,576,4096] p-major (raw-reshape identity),
//                    masked WMMA GEMM vs W2 + bias -> out [16,128,64,64]
// ---------------------------------------------------------------------------

typedef __attribute__((ext_vector_type(16))) __bf16 v16b;
typedef __attribute__((ext_vector_type(8)))  __bf16 v8b;
typedef __attribute__((ext_vector_type(8)))  float  v8f;

// ---- workspace layout (bytes) ---------------------------------------------
#define OFF_FLAT32 (0ull)
#define OFF_FINAL  (150994944ull)                    // + N*576*4
#define OFF_FLATB  (OFF_FINAL + 150994944ull)        // + N*576*4
#define OFF_RN     (OFF_FLATB + 75497472ull)         // + N*576*2
#define OFF_CCB    (OFF_RN + 262144ull)              // + N*4
#define OFF_CCT    (OFF_CCB + 589824ull)             // + 512*576*2
#define OFF_CN     (OFF_CCT + 589824ull)             // + 576*512*2
#define OFF_W2     (OFF_CN + 2048ull)                // + 512*4
// total ~362 MiB

// ---- WMMA fragment loaders (bf16 16x16x32, ISA 7.12.2 layouts) -------------
// A matrix 16x32: lane = h*16 + r holds row r, K runs {h*8..h*8+7} and {16+h*8..}
template <typename P>
__device__ __forceinline__ v16b ldA(P base, int stride, int row0, int kB, int lane) {
  const int r = lane & 15;
  const int h = (lane >> 4) & 1;
  auto p = base + (size_t)(row0 + r) * stride + kB + h * 8;
  v8b lo = *(const v8b*)(p);
  v8b hi = *(const v8b*)(p + 16);
  return __builtin_shufflevector(lo, hi, 0, 1, 2, 3, 4, 5, 6, 7,
                                         8, 9, 10, 11, 12, 13, 14, 15);
}

// B matrix 32x16: lane = h*16 + c holds column c, K runs {kB+h*8..kB+h*8+7} and {kB+16+h*8..}
// (B stored row-major [N, K], so a column of B == a contiguous row here)
template <typename P>
__device__ __forceinline__ v16b ldB(P base, int stride, int col0, int kB, int lane) {
  const int c = lane & 15;
  const int h = (lane >> 4) & 1;
  auto p = base + (size_t)(col0 + c) * stride + kB + h * 8;
  v8b lo = *(const v8b*)(p);
  v8b hi = *(const v8b*)(p + 16);
  return __builtin_shufflevector(lo, hi, 0, 1, 2, 3, 4, 5, 6, 7,
                                         8, 9, 10, 11, 12, 13, 14, 15);
}

__device__ __forceinline__ v8f wmma_bf16(v16b a, v16b b, v8f c) {
  return __builtin_amdgcn_wmma_f32_16x16x32_bf16(false, a, false, b,
                                                 (short)0, c, false, false);
}

// ---- K1: im2col + norms ---------------------------------------------------
__global__ __launch_bounds__(256) void im2col_kernel(
    const float* __restrict__ x, float* __restrict__ flat32,
    __bf16* __restrict__ flatB, float* __restrict__ rn) {
  const int tid = threadIdx.x;
  const int b = blockIdx.x >> 6;
  const int h = blockIdx.x & 63;
  __shared__ float rnLoc[64];
  if (tid < 64) rnLoc[tid] = 0.f;
  __syncthreads();
  for (int idx = tid; idx < 64 * 576; idx += 256) {
    const int w = idx / 576;
    const int p = idx - w * 576;
    const int c = p / 9;
    const int q = p - c * 9;
    const int kh = q / 3;
    const int kw = q - kh * 3;
    const int hh = h + kh - 1, ww = w + kw - 1;
    float v = 0.f;
    if (hh >= 0 && hh < 64 && ww >= 0 && ww < 64)
      v = x[(((size_t)b * 64 + c) * 64 + hh) * 64 + ww];
    const size_t row = (size_t)blockIdx.x * 64 + w;
    flat32[row * 576 + p] = v;
    flatB[row * 576 + p] = (__bf16)v;
    atomicAdd(&rnLoc[w], v * v);
  }
  __syncthreads();
  if (tid < 64) rn[(size_t)blockIdx.x * 64 + tid] = rnLoc[tid];
}

// ---- K2: centers -> bf16 row-major + bf16 transpose + norms ---------------
__global__ __launch_bounds__(256) void prep_centers_kernel(
    const float* __restrict__ cc, __bf16* __restrict__ ccB,
    __bf16* __restrict__ ccT, float* __restrict__ cn) {
  const int n = blockIdx.x;
  const int tid = threadIdx.x;
  __shared__ float acc;
  if (tid == 0) acc = 0.f;
  __syncthreads();
  float part = 0.f;
  for (int j = tid; j < 576; j += 256) {
    const float v = cc[(size_t)n * 576 + j];
    ccB[(size_t)n * 576 + j] = (__bf16)v;
    ccT[(size_t)j * 512 + n] = (__bf16)v;
    part += v * v;
  }
  atomicAdd(&acc, part);
  __syncthreads();
  if (tid == 0) cn[n] = acc;
}

// ---- K2w: weights -> bf16 (layout [O, C*K*K] already matches p-order) -----
__global__ __launch_bounds__(256) void prep_weights_kernel(
    const float* __restrict__ w, __bf16* __restrict__ W2) {
  const int idx = blockIdx.x * 256 + threadIdx.x;  // 128*576 = 73728 elems
  W2[idx] = (__bf16)w[idx];
}

// ---- K3: fused GEMM1 -> sqrt-softmax -> GEMM2 -> blend --------------------
__global__ __launch_bounds__(256) void gemm_fused_kernel(
    const __bf16* __restrict__ flatB, const float* __restrict__ flat32,
    const __bf16* __restrict__ ccB, const __bf16* __restrict__ ccT,
    const float* __restrict__ rn, const float* __restrict__ cn,
    const float* __restrict__ temp, float* __restrict__ finalBuf) {
  __shared__ __align__(16) __bf16 ldsA[16 * 576];   // A tile, then probs [16][512]
  __shared__ float ldsS[16 * 512];                  // scores / logits / exp
  const int tid = threadIdx.x;
  const int lane = tid & 31;
  const int wave = tid >> 5;                        // 8 waves (wave32)
  const size_t rowBase = (size_t)blockIdx.x * 16;

  // stage A tile (16x576 bf16 = 18 KB) into LDS
  {
    const uint4* src = (const uint4*)(flatB + rowBase * 576);
    uint4* dst = (uint4*)ldsA;
    for (int i = tid; i < 1152; i += 256) dst[i] = src[i];
  }
  __syncthreads();

  const float tv = temp[0];

  // ---- GEMM1: scores[16][512] = A(16x576) * ccB^T, K-loop of 18 WMMAs ----
  for (int s = 0; s < 4; ++s) {
    const int nBase = (wave * 4 + s) * 16;
    __builtin_prefetch(ccB + (size_t)nBase * 576, 0, 1);   // L2-resident B panel
    v8f acc = {0.f, 0.f, 0.f, 0.f, 0.f, 0.f, 0.f, 0.f};
    for (int kt = 0; kt < 18; ++kt) {
      const int kB = kt * 32;
      v16b a = ldA(ldsA, 576, 0, kB, lane);
      v16b bm = ldB(ccB, 576, nBase, kB, lane);
      acc = wmma_bf16(a, bm, acc);
    }
    const int n = nBase + (lane & 15);
    const int h = (lane >> 4) & 1;
    for (int r = 0; r < 8; ++r) ldsS[(r + 8 * h) * 512 + n] = acc[r];
  }
  __syncthreads();

  // ---- sqrt-distance softmax over 512, 16 lanes per row (wave32 shfl) ----
  {
    const int row = tid >> 4;
    const int c0 = (tid & 15) * 32;
    const float rnv = rn[rowBase + row];
    float mx = -3.0e38f;
    for (int c = c0; c < c0 + 32; ++c) {
      const float sdot = ldsS[row * 512 + c];
      const float d2 = rnv + cn[c] - 2.f * sdot;
      const float l = -sqrtf(fmaxf(d2, 0.f)) * tv;
      ldsS[row * 512 + c] = l;
      mx = fmaxf(mx, l);
    }
    for (int m = 1; m < 16; m <<= 1) mx = fmaxf(mx, __shfl_xor(mx, m, 32));
    float sum = 0.f;
    for (int c = c0; c < c0 + 32; ++c) {
      const float e = __expf(ldsS[row * 512 + c] - mx);
      ldsS[row * 512 + c] = e;
      sum += e;
    }
    for (int m = 1; m < 16; m <<= 1) sum += __shfl_xor(sum, m, 32);
    const float inv = 1.f / sum;
    __bf16* probs = ldsA;                       // reuse A-tile region (16 KB)
    for (int c = c0; c < c0 + 32; ++c)
      probs[row * 512 + c] = (__bf16)(ldsS[row * 512 + c] * inv);
  }
  __syncthreads();

  // ---- GEMM2: transformed(16x576) = probs(16x512) * cc, then blend -------
  const float invT1 = 1.f / (tv + 1.f);
  for (int jSub = wave; jSub < 36; jSub += 8) {
    const int jBase = jSub * 16;
    __builtin_prefetch(ccT + (size_t)jBase * 512, 0, 1);
    v8f acc = {0.f, 0.f, 0.f, 0.f, 0.f, 0.f, 0.f, 0.f};
    for (int kt = 0; kt < 16; ++kt) {
      const int kB = kt * 32;
      v16b a = ldA(ldsA, 512, 0, kB, lane);     // probs as A matrix
      v16b bm = ldB(ccT, 512, jBase, kB, lane);
      acc = wmma_bf16(a, bm, acc);
    }
    const int j = jBase + (lane & 15);
    const int h = (lane >> 4) & 1;
    for (int r = 0; r < 8; ++r) {
      const size_t row = rowBase + r + 8 * h;
      const size_t off = row * 576 + j;
      finalBuf[off] = (tv * acc[r] + flat32[off]) * invT1;
    }
  }
}

// ---- K4: fold/crop/conv tail == masked GEMM on final viewed [B,576,4096] --
__global__ __launch_bounds__(256) void conv_out_kernel(
    const float* __restrict__ finalBuf, const __bf16* __restrict__ W2,
    const float* __restrict__ bias, float* __restrict__ out) {
  __shared__ __align__(16) __bf16 G[32 * 576];     // 32 pixels x 576 feats
  const int blk = blockIdx.x;                      // b*128 + i*2 + jh
  const int b = blk >> 7;
  const int rem = blk & 127;
  const int i = rem >> 1;
  const int jh = rem & 1;
  const int jBase = jh * 32;
  const int tid = threadIdx.x;
  const int lane = tid & 31;
  const int wave = tid >> 5;

  // gather: coalesced along pixels (consecutive l), mask dropped taps
  for (int idx = tid; idx < 32 * 576; idx += 256) {
    const int m = idx & 31;
    const int p = idx >> 5;
    float v = finalBuf[((size_t)b * 576 + p) * 4096 + (size_t)i * 64 + jBase + m];
    const int q = p % 9;
    const int u = q / 3;
    const int vk = q - u * 3;
    if ((i == 0 && u == 0) || (jBase + m == 0 && vk == 0)) v = 0.f;
    G[m * 576 + p] = (__bf16)v;
  }
  __syncthreads();

  // 2 m-subtiles x 8 o-subtiles = 16 WMMA tiles over 8 waves
  for (int t = wave; t < 16; t += 8) {
    const int mSub = t & 1;
    const int nSub = t >> 1;
    __builtin_prefetch(W2 + (size_t)nSub * 16 * 576, 0, 1);
    v8f acc = {0.f, 0.f, 0.f, 0.f, 0.f, 0.f, 0.f, 0.f};
    for (int kt = 0; kt < 18; ++kt) {
      const int kB = kt * 32;
      v16b a = ldA(G, 576, mSub * 16, kB, lane);
      v16b bm = ldB(W2, 576, nSub * 16, kB, lane);
      acc = wmma_bf16(a, bm, acc);
    }
    const int o = nSub * 16 + (lane & 15);
    const float bo = bias[o];
    const int h = (lane >> 4) & 1;
    for (int r = 0; r < 8; ++r) {
      const int m = mSub * 16 + 8 * h + r;       // pixel within 32-tile
      out[(((size_t)b * 128 + o) * 64 + i) * 64 + jBase + m] = acc[r] + bo;
    }
  }
}

// ---------------------------------------------------------------------------
extern "C" void kernel_launch(void* const* d_in, const int* in_sizes, int n_in,
                              void* d_out, int out_size, void* d_ws, size_t ws_size,
                              hipStream_t stream) {
  const float* x      = (const float*)d_in[0];   // [16,64,64,64]
  const float* cc     = (const float*)d_in[1];   // [512,576]
  const float* temp   = (const float*)d_in[2];   // [1]
  const float* weight = (const float*)d_in[3];   // [128,64,3,3]
  const float* bias   = (const float*)d_in[4];   // [128]
  float* out = (float*)d_out;                    // [16,128,64,64]

  char* ws = (char*)d_ws;
  float*  flat32   = (float*)(ws + OFF_FLAT32);
  float*  finalBuf = (float*)(ws + OFF_FINAL);
  __bf16* flatB    = (__bf16*)(ws + OFF_FLATB);
  float*  rnBuf    = (float*)(ws + OFF_RN);
  __bf16* ccB      = (__bf16*)(ws + OFF_CCB);
  __bf16* ccT      = (__bf16*)(ws + OFF_CCT);
  float*  cnBuf    = (float*)(ws + OFF_CN);
  __bf16* W2       = (__bf16*)(ws + OFF_W2);

  prep_centers_kernel<<<512, 256, 0, stream>>>(cc, ccB, ccT, cnBuf);
  prep_weights_kernel<<<288, 256, 0, stream>>>(weight, W2);
  im2col_kernel<<<1024, 256, 0, stream>>>(x, flat32, flatB, rnBuf);
  gemm_fused_kernel<<<4096, 256, 0, stream>>>(flatB, flat32, ccB, ccT,
                                              rnBuf, cnBuf, temp, finalBuf);
  conv_out_kernel<<<2048, 256, 0, stream>>>(finalBuf, W2, bias, out);
}
